// MambaBlock_76166950027467
// MI455X (gfx1250) — matrix operand order repaired and checked by reading, not compiled, not run on hardware
//
#include <hip/hip_runtime.h>
#include <math.h>

#define DIM_     1024
#define D_STATE_ 16
#define D_CONV_  4
#define DT_RANK_ 64
#define D_INNER_ 2048
#define BATCH_   2
#define SEQ_     1024
#define M_TOT    (BATCH_ * SEQ_)            // 2048
#define XPN_     (DT_RANK_ + 2 * D_STATE_)  // 96
#define DEPTH_   2

typedef __attribute__((ext_vector_type(16))) _Float16 v16h;
typedef __attribute__((ext_vector_type(8)))  _Float16 h8;
typedef __attribute__((ext_vector_type(4)))  _Float16 h4;
typedef __attribute__((ext_vector_type(8)))  float    v8f;
typedef __attribute__((ext_vector_type(4)))  float    f4;
typedef int i32x4_ __attribute__((vector_size(16)));

#define TILE_M 128
#define TILE_N 128
#define TILE_K 32
#define LDS_STRIDE 40   // halves per row: pad, keep 16B alignment

// --------------------------------------------------------------------------
// Async global->LDS copy (CDNA5 GLOBAL_LOAD_ASYNC_TO_LDS_B128, ASYNCcnt) with
// a synchronous b128 fallback if the builtin is absent in this toolchain.
// --------------------------------------------------------------------------
#if defined(__HIP_DEVICE_COMPILE__) && defined(__has_builtin)
#if __has_builtin(__builtin_amdgcn_global_load_async_to_lds_b128)
#define USE_ASYNC_LDS 1
#endif
#endif

#define AS_G __attribute__((address_space(1)))
#define AS_L __attribute__((address_space(3)))

__device__ __forceinline__ void copy16_g2l(const _Float16* g, _Float16* l) {
#ifdef USE_ASYNC_LDS
  __builtin_amdgcn_global_load_async_to_lds_b128(
      (AS_G i32x4_*)g, (AS_L i32x4_*)l, 0, 0);
#else
  *(h8*)l = *(const h8*)g;
#endif
}

__device__ __forceinline__ void wait_async_copies() {
#ifdef USE_ASYNC_LDS
#if __has_builtin(__builtin_amdgcn_s_wait_asynccnt)
  __builtin_amdgcn_s_wait_asynccnt(0);
#else
  asm volatile("s_wait_asynccnt 0x0" ::: "memory");
#endif
#endif
}

// --------------------------------------------------------------------------
// C[M,N] = A[M,K] * B[N,K]^T, A/B f16, C f32. Block tile 128x128, 8 waves
// (4Mx2N), wave tile 32x64 -> 8 v_wmma_f32_16x16x32_f16 per K-step.
// Grid: (ceil(N/128), M/128). M % 128 == 0, K % 32 == 0. N guarded
// (OOB B rows clamped on load, C stores guarded).
// --------------------------------------------------------------------------
__global__ __launch_bounds__(256) void gemm_nt_wmma_f16(
    const _Float16* __restrict__ A, int lda,
    const _Float16* __restrict__ Bw, int ldb,
    float* __restrict__ C, int ldc,
    int N, int K)
{
  __shared__ _Float16 lA[TILE_M * LDS_STRIDE];
  __shared__ _Float16 lB[TILE_N * LDS_STRIDE];

  const int tid    = threadIdx.x;
  const int lane   = tid & 31;
  const int wave   = tid >> 5;
  const int blockM = blockIdx.y * TILE_M;
  const int blockN = blockIdx.x * TILE_N;
  const int waveM  = (wave & 3) * 32;   // 4 waves along M
  const int waveN  = (wave >> 2) * 64;  // 2 waves along N

  v8f acc[2][4];
#pragma unroll
  for (int i = 0; i < 2; ++i)
#pragma unroll
    for (int j = 0; j < 4; ++j)
#pragma unroll
      for (int e = 0; e < 8; ++e) acc[i][j][e] = 0.0f;

  const int mrow = lane & 15;
  const int kA   = (lane >> 4) * 8;    // A frag K base (ISA 16-bit A layout)
  const int kB   = (lane >> 4) * 8;    // B frag K base (same K order as A)

  for (int k0 = 0; k0 < K; k0 += TILE_K) {
    __syncthreads();
    // ---- stage A tile: 128 rows x 32 halves = 512 b128 chunks (2/thread)
#pragma unroll
    for (int i = 0; i < 2; ++i) {
      int chunk = tid + i * 256;
      int row   = chunk >> 2;
      int cs    = (chunk & 3) * 8;
      copy16_g2l(A + (size_t)(blockM + row) * lda + k0 + cs,
                 &lA[row * LDS_STRIDE + cs]);
    }
    // ---- stage B tile: 128 rows x 32 halves (OOB rows clamped: finite
    //      garbage in never-stored columns)
#pragma unroll
    for (int i = 0; i < 2; ++i) {
      int chunk = tid + i * 256;
      int row   = chunk >> 2;
      int cs    = (chunk & 3) * 8;
      int n     = blockN + row;
      if (n >= N) n = N - 1;
      copy16_g2l(Bw + (size_t)n * ldb + k0 + cs,
                 &lB[row * LDS_STRIDE + cs]);
    }
    wait_async_copies();
    __syncthreads();

    // ---- fragments (two b128 LDS loads each) + 8 WMMAs per wave
    union Frag { v16h v; h8 h[2]; };
    Frag a[2], b[4];
#pragma unroll
    for (int i = 0; i < 2; ++i) {
      int r = waveM + i * 16 + mrow;
      a[i].h[0] = *(const h8*)&lA[r * LDS_STRIDE + kA];
      a[i].h[1] = *(const h8*)&lA[r * LDS_STRIDE + kA + 16];
    }
#pragma unroll
    for (int j = 0; j < 4; ++j) {
      int r = waveN + j * 16 + mrow;
      b[j].h[0] = *(const h8*)&lB[r * LDS_STRIDE + kB];
      b[j].h[1] = *(const h8*)&lB[r * LDS_STRIDE + kB + 16];
    }
#pragma unroll
    for (int i = 0; i < 2; ++i)
#pragma unroll
      for (int j = 0; j < 4; ++j)
        acc[i][j] = __builtin_amdgcn_wmma_f32_16x16x32_f16(
            false, a[i].v, false, b[j].v, (short)0, acc[i][j], false, false);
  }

  // ---- store C (f32 C/D layout: lane&15 = N col, (lane>>4)*8 + r = M row)
  const int nBase = blockN + waveN + (lane & 15);
  const int rBase = blockM + waveM + (lane >> 4) * 8;
#pragma unroll
  for (int i = 0; i < 2; ++i)
#pragma unroll
    for (int j = 0; j < 4; ++j) {
      int n = nBase + j * 16;
      if (n < N) {
#pragma unroll
        for (int r = 0; r < 8; ++r)
          C[(size_t)(rBase + i * 16 + r) * ldc + n] = acc[i][j][r];
      }
    }
}

// --------------------------------------------------------------------------
// fp32 -> f16 conversion (n % 1024 == 0), 4 elems/thread vectorized
// --------------------------------------------------------------------------
__global__ __launch_bounds__(256) void cvt_f16_kernel(
    const float* __restrict__ src, _Float16* __restrict__ dst)
{
  int idx = (blockIdx.x * 256 + threadIdx.x) * 4;
  f4 v = *(const f4*)(src + idx);
  h4 hv;
  hv.x = (_Float16)v.x; hv.y = (_Float16)v.y;
  hv.z = (_Float16)v.z; hv.w = (_Float16)v.w;
  *(h4*)(dst + idx) = hv;
}

// --------------------------------------------------------------------------
// Depthwise causal conv (D_CONV=4) + bias + silu; writes f16 activations.
// xz: [B,L,2*D_INNER] fp32, xi = xz[...,:D_INNER]. One thread per (b,l,d).
// --------------------------------------------------------------------------
__global__ __launch_bounds__(256) void conv_silu_kernel(
    const float* __restrict__ xz, const float* __restrict__ cw,
    const float* __restrict__ cb, _Float16* __restrict__ xch)
{
  int idx = blockIdx.x * 256 + threadIdx.x;       // < M_TOT * D_INNER
  int d  = idx & (D_INNER_ - 1);
  int bl = idx >> 11;                             // / D_INNER
  int l  = bl & (SEQ_ - 1);
  float accv = cb[d];
#pragma unroll
  for (int j = 0; j < D_CONV_; ++j) {
    int ls = l - (D_CONV_ - 1) + j;
    if (ls >= 0)
      accv += cw[d * D_CONV_ + j] *
              xz[((size_t)bl + (ls - l)) * (2 * D_INNER_) + d];
  }
  xch[(size_t)bl * D_INNER_ + d] =
      (_Float16)(accv / (1.0f + __expf(-accv)));
}

// --------------------------------------------------------------------------
// dt = softplus(dt_lin + dtb[d])  (in place, fp32)
// --------------------------------------------------------------------------
__global__ __launch_bounds__(256) void softplus_bias_kernel(
    float* __restrict__ dt, const float* __restrict__ dtb)
{
  int idx = blockIdx.x * 256 + threadIdx.x;
  int d = idx & (D_INNER_ - 1);
  float v = dt[idx] + dtb[d];
  dt[idx] = (v > 20.0f) ? v : log1pf(__expf(v));
}

// --------------------------------------------------------------------------
// Selective scan + skip (D*u) + silu(z) gate. One thread per (b,d); h[16] in
// registers; B/C vectors broadcast from x_dbl. Writes f16 y for out_proj.
// --------------------------------------------------------------------------
__global__ __launch_bounds__(256) void scan_kernel(
    const float* __restrict__ dt, const float* __restrict__ xdbl,
    const _Float16* __restrict__ xch, const float* __restrict__ xz,
    const float* __restrict__ A_log, const float* __restrict__ Dp,
    _Float16* __restrict__ yh)
{
  int t = blockIdx.x * 256 + threadIdx.x;   // < BATCH_*D_INNER_
  int d = t & (D_INNER_ - 1);
  int b = t >> 11;
  float a[D_STATE_], h[D_STATE_];
#pragma unroll
  for (int n = 0; n < D_STATE_; ++n) {
    a[n] = -__expf(A_log[d * D_STATE_ + n]);
    h[n] = 0.0f;
  }
  float Dd = Dp[d];
  for (int l = 0; l < SEQ_; ++l) {
    size_t bl = (size_t)b * SEQ_ + l;
    float dtv = dt[bl * D_INNER_ + d];
    float u   = (float)xch[bl * D_INNER_ + d];
    float zv  = xz[bl * (2 * D_INNER_) + D_INNER_ + d];
    const float* Bp = xdbl + bl * XPN_ + DT_RANK_;
    const float* Cp = Bp + D_STATE_;
    float yv = 0.0f;
#pragma unroll
    for (int n = 0; n < D_STATE_; ++n) {
      float dA = __expf(dtv * a[n]);
      h[n] = dA * h[n] + (dtv * Bp[n]) * u;
      yv += h[n] * Cp[n];
    }
    yv += Dd * u;
    yv *= zv / (1.0f + __expf(-zv));     // y * silu(z)
    yh[bl * D_INNER_ + d] = (_Float16)yv;
  }
}

// --------------------------------------------------------------------------
extern "C" void kernel_launch(void* const* d_in, const int* in_sizes, int n_in,
                              void* d_out, int out_size, void* d_ws, size_t ws_size,
                              hipStream_t stream) {
  (void)in_sizes; (void)n_in; (void)out_size; (void)ws_size;
  const float* x     = (const float*)d_in[0];
  const float* in_w  = (const float*)d_in[1];   // [DEPTH, 4096, 1024]
  const float* cw    = (const float*)d_in[2];   // [DEPTH, 2048, 4]
  const float* cb    = (const float*)d_in[3];   // [DEPTH, 2048]
  const float* xp_w  = (const float*)d_in[4];   // [DEPTH, 96, 2048]
  const float* dtw   = (const float*)d_in[5];   // [DEPTH, 2048, 64]
  const float* dtb   = (const float*)d_in[6];   // [DEPTH, 2048]
  const float* A_log = (const float*)d_in[7];   // [DEPTH, 2048, 16]
  const float* Dp    = (const float*)d_in[8];   // [DEPTH, 2048]
  const float* ow    = (const float*)d_in[9];   // [DEPTH, 1024, 2048]

  const size_t M = M_TOT;
  // fp32 scratch
  float* xz  = (float*)d_ws;                    // M*4096
  float* xdb = xz  + M * 4096;                  // M*96
  float* dt  = xdb + M * XPN_;                  // M*2048
  float* cur = dt  + M * D_INNER_;              // M*1024 (layer0 output)
  // f16 scratch
  _Float16* xh   = (_Float16*)(cur + M * DIM_); // M*1024  layer input
  _Float16* xch  = xh   + M * DIM_;             // M*2048  conv+silu out
  _Float16* xdbh = xch  + M * D_INNER_;         // M*96    dt_raw
  _Float16* yh   = xdbh + M * XPN_;             // M*2048  scan out
  _Float16* whA  = yh   + M * D_INNER_;         // 4096*1024 in_proj_w
  _Float16* whB  = whA + (size_t)(2 * D_INNER_) * DIM_;   // 96*2048
  _Float16* whC  = whB + (size_t)XPN_ * D_INNER_;         // 2048*64
  _Float16* whD  = whC + (size_t)D_INNER_ * DT_RANK_;     // 1024*2048

  const dim3 blk(256);
  for (int layer = 0; layer < DEPTH_; ++layer) {
    const float* lin   = (layer == 0) ? x : cur;
    float* layer_out   = (layer == DEPTH_ - 1) ? (float*)d_out : cur;
    const float* inw_l = in_w  + (size_t)layer * (2 * D_INNER_) * DIM_;
    const float* cw_l  = cw    + (size_t)layer * D_INNER_ * D_CONV_;
    const float* cb_l  = cb    + (size_t)layer * D_INNER_;
    const float* xpw_l = xp_w  + (size_t)layer * XPN_ * D_INNER_;
    const float* dtw_l = dtw   + (size_t)layer * D_INNER_ * DT_RANK_;
    const float* dtb_l = dtb   + (size_t)layer * D_INNER_;
    const float* Al_l  = A_log + (size_t)layer * D_INNER_ * D_STATE_;
    const float* Dp_l  = Dp    + (size_t)layer * D_INNER_;
    const float* ow_l  = ow    + (size_t)layer * DIM_ * D_INNER_;

    // f16 conversions: layer input + this layer's weights
    cvt_f16_kernel<<<(M * DIM_) / 1024, blk, 0, stream>>>(lin, xh);
    cvt_f16_kernel<<<((size_t)(2 * D_INNER_) * DIM_) / 1024, blk, 0, stream>>>(inw_l, whA);
    cvt_f16_kernel<<<((size_t)XPN_ * D_INNER_) / 1024, blk, 0, stream>>>(xpw_l, whB);
    cvt_f16_kernel<<<((size_t)D_INNER_ * DT_RANK_) / 1024, blk, 0, stream>>>(dtw_l, whC);
    cvt_f16_kernel<<<((size_t)DIM_ * D_INNER_) / 1024, blk, 0, stream>>>(ow_l, whD);

    // 1) in_proj: xz = x * in_w^T   [2048 x 4096, K=1024]
    gemm_nt_wmma_f16<<<dim3((2 * D_INNER_) / TILE_N, M / TILE_M), blk, 0, stream>>>(
        xh, DIM_, whA, DIM_, xz, 2 * D_INNER_, 2 * D_INNER_, DIM_);
    // 2) depthwise conv + silu -> xch (f16)
    conv_silu_kernel<<<(M * D_INNER_) / 256, blk, 0, stream>>>(xz, cw_l, cb_l, xch);
    // 3) x_proj: xdb = xc * xp_w^T   [2048 x 96, K=2048]
    gemm_nt_wmma_f16<<<dim3((XPN_ + TILE_N - 1) / TILE_N, M / TILE_M), blk, 0, stream>>>(
        xch, D_INNER_, whB, D_INNER_, xdb, XPN_, XPN_, D_INNER_);
    // 3b) dt_raw -> f16 (whole xdb row incl. B/C cols; GEMM reads first 64)
    cvt_f16_kernel<<<(M * XPN_) / 1024, blk, 0, stream>>>(xdb, xdbh);
    // 4) dt_proj: dt = dt_raw * dtw^T [2048 x 2048, K=64] (lda = 96)
    gemm_nt_wmma_f16<<<dim3(D_INNER_ / TILE_N, M / TILE_M), blk, 0, stream>>>(
        xdbh, XPN_, whC, DT_RANK_, dt, D_INNER_, D_INNER_, DT_RANK_);
    // 5) dt = softplus(dt + dtb)
    softplus_bias_kernel<<<(M * D_INNER_) / 256, blk, 0, stream>>>(dt, dtb_l);
    // 6) selective scan + skip + gate -> yh (f16)
    scan_kernel<<<(BATCH_ * D_INNER_) / 256, blk, 0, stream>>>(
        dt, xdb, xch, xz, Al_l, Dp_l, yh);
    // 7) out_proj: out = y * ow^T    [2048 x 1024, K=2048]
    gemm_nt_wmma_f16<<<dim3(DIM_ / TILE_N, M / TILE_M), blk, 0, stream>>>(
        yh, D_INNER_, whD, D_INNER_, layer_out, DIM_, DIM_, D_INNER_);
  }
}
